// MultiHeadSelfAttention_45389214384452
// MI455X (gfx1250) — matrix operand order repaired and checked by reading, not compiled, not run on hardware
//
#include <hip/hip_runtime.h>
#include <math.h>

// Problem constants (match reference)
#define B_   2
#define S_   2048
#define D_   2048
#define H_   16
#define HD_  128
#define N3D  (3 * D_)              // 6144
#define SCALE_Q 0.08838834764831845f   // 128^-0.5

typedef __attribute__((ext_vector_type(16))) __bf16 v16bf;
typedef __attribute__((ext_vector_type(8)))  float  v8f;
typedef unsigned short bfraw;   // bf16 storage type (host-safe)

// ---------- bf16 helpers (bit-punned, round-to-nearest-even) ----------
__device__ __forceinline__ bfraw f2bf_raw(float f) {
  union { float f; unsigned u; } v; v.f = f;
  unsigned r = (v.u + 0x7FFFu + ((v.u >> 16) & 1u)) >> 16;
  return (bfraw)r;
}
__device__ __forceinline__ __bf16 raw2bf(bfraw r) {
  union { bfraw s; __bf16 b; } u; u.s = r; return u.b;
}

// ---------- WMMA fragment loaders (wave32, 16x16x32 bf16) ----------
// A fragment (16 rows x 32 K): lane L holds row m0 + (L&15);
// element e -> K = e + 16*(L>>4)
__device__ __forceinline__ v16bf load_a_frag(const bfraw* __restrict__ base,
                                             long row_stride, int m0, int k0, int lane) {
  const int half = lane >> 4;
  const bfraw* p = base + (long)(m0 + (lane & 15)) * row_stride + k0 + 16 * half;
  v16bf a;
#pragma unroll
  for (int e = 0; e < 16; ++e) a[e] = raw2bf(p[e]);
  return a;
}

// B fragment (32 K x 16 cols), source stored row-major as [col][K]
// (i.e. weight W[N,K], or K-cache [key][hd]): lane L holds col n0 + (L&15);
// element e -> K = e + 16*(L>>4)   -> contiguous 32B read per lane
__device__ __forceinline__ v16bf load_b_frag_rowmajNK(const bfraw* __restrict__ base,
                                                      long row_stride, int n0, int k0, int lane) {
  const int half = lane >> 4;
  const bfraw* p = base + (long)(n0 + (lane & 15)) * row_stride + k0 + 16 * half;
  v16bf b;
#pragma unroll
  for (int e = 0; e < 16; ++e) b[e] = raw2bf(p[e]);
  return b;
}

__device__ __forceinline__ v8f wmma_bf16(v16bf a, v16bf b, v8f c) {
  return __builtin_amdgcn_wmma_f32_16x16x32_bf16(false, a, false, b, (short)0, c, false, false);
}

#define V8F_ZERO {0.f,0.f,0.f,0.f,0.f,0.f,0.f,0.f}

// ---------- 1) f32 -> bf16 conversion ----------
__global__ void cvt_f32_bf16(const float* __restrict__ in, bfraw* __restrict__ out, long n) {
  long i = (long)blockIdx.x * blockDim.x + threadIdx.x;
  long stride = (long)gridDim.x * blockDim.x;
  for (; i < n; i += stride) out[i] = f2bf_raw(in[i]);
}

// ---------- 2) QKV GEMM: [B*S,D] x [3D,D]^T + bias, scatter to head-major Q/K/V ----------
// grid: x = N3D/256, y = (B*S)/16 ; block = 128 (4 waves), wave tile = 16 x 64
__global__ __launch_bounds__(128) void qkv_gemm(const bfraw* __restrict__ X,
                                                const bfraw* __restrict__ W,
                                                const float* __restrict__ bias,
                                                bfraw* __restrict__ Q,
                                                bfraw* __restrict__ Kc,
                                                bfraw* __restrict__ V) {
  const int lane = threadIdx.x & 31;
  const int wave = threadIdx.x >> 5;
  const int half = lane >> 4;
  const int col  = lane & 15;
  const int m0   = blockIdx.y * 16;
  const int nw0  = (blockIdx.x * 4 + wave) * 64;

  v8f acc[4] = {V8F_ZERO, V8F_ZERO, V8F_ZERO, V8F_ZERO};
  for (int k0 = 0; k0 < D_; k0 += 32) {
    if (k0 + 32 < D_)
      __builtin_prefetch((const void*)(X + (long)(m0 + col) * D_ + k0 + 32), 0, 0);
    v16bf a = load_a_frag(X, D_, m0, k0, lane);
#pragma unroll
    for (int j = 0; j < 4; ++j) {
      v16bf b = load_b_frag_rowmajNK(W, D_, nw0 + j * 16, k0, lane);
      acc[j] = wmma_bf16(a, b, acc[j]);
    }
  }
#pragma unroll
  for (int j = 0; j < 4; ++j) {
    const int n = nw0 + j * 16 + col;
    const float bs = bias[n];
    const int which = n / D_;
    const int rem = n - which * D_;
    const int h = rem / HD_;
    const int hd = rem - h * HD_;
    bfraw* dst = (which == 0) ? Q : (which == 1) ? Kc : V;
#pragma unroll
    for (int r = 0; r < 8; ++r) {
      const int m = m0 + r + 8 * half;        // token index in [0, B*S)
      const int bb = m / S_;
      const int s = m - bb * S_;
      float v = acc[j][r] + bs;
      if (which == 0) v *= SCALE_Q;
      dst[(((long)bb * H_ + h) * S_ + s) * HD_ + hd] = f2bf_raw(v);
    }
  }
}

// ---------- 3) Flash attention: one wave per (b,h, 16 q rows) ----------
// grid: x = S/16, y = B*H ; block = 32
__global__ __launch_bounds__(32) void flash_attn(const bfraw* __restrict__ Q,
                                                 const bfraw* __restrict__ Kc,
                                                 const bfraw* __restrict__ V,
                                                 const float* __restrict__ amask,
                                                 const unsigned char* __restrict__ kpad,
                                                 bfraw* __restrict__ O) {
  __shared__ bfraw ps[16 * 32];   // P tile staging: D-layout -> A-fragment layout
  const int lane = threadIdx.x & 31;
  const int half = lane >> 4;
  const int col  = lane & 15;
  const int bh = blockIdx.y;
  const int b = bh / H_;
  const int h = bh - b * H_;
  const int m0 = blockIdx.x * 16;

  const bfraw* Qb = Q  + (long)bh * S_ * HD_;
  const bfraw* Kb = Kc + (long)bh * S_ * HD_;
  const bfraw* Vb = V  + (long)bh * S_ * HD_;

  v16bf qf[4];
#pragma unroll
  for (int f = 0; f < 4; ++f) qf[f] = load_a_frag(Qb, HD_, m0, f * 32, lane);

  v8f o[8] = {V8F_ZERO, V8F_ZERO, V8F_ZERO, V8F_ZERO,
              V8F_ZERO, V8F_ZERO, V8F_ZERO, V8F_ZERO};
  float rm[8], rl[8];
#pragma unroll
  for (int r = 0; r < 8; ++r) { rm[r] = -__builtin_inff(); rl[r] = 0.f; }

  for (int kt = 0; kt < S_; kt += 32) {
    // S = Q . K^T for 32 keys (two 16-col tiles)
    v8f sa[2] = {V8F_ZERO, V8F_ZERO};
#pragma unroll
    for (int c = 0; c < 2; ++c)
#pragma unroll
      for (int f = 0; f < 4; ++f) {
        v16bf kf = load_b_frag_rowmajNK(Kb, HD_, kt + c * 16, f * 32, lane);
        sa[c] = wmma_bf16(qf[f], kf, sa[c]);
      }

    // masks + online softmax (row stats live replicated in each 16-lane half)
    const float kadd0 = kpad[b * S_ + kt + col]      ? -__builtin_inff() : 0.f;
    const float kadd1 = kpad[b * S_ + kt + 16 + col] ? -__builtin_inff() : 0.f;
    float p0v[8], p1v[8], alpha[8];
#pragma unroll
    for (int r = 0; r < 8; ++r) {
      const int m = m0 + r + 8 * half;
      float s0 = sa[0][r] + amask[(long)m * S_ + kt + col]      + kadd0;
      float s1 = sa[1][r] + amask[(long)m * S_ + kt + 16 + col] + kadd1;
      float tm = fmaxf(s0, s1);
#pragma unroll
      for (int msk = 1; msk < 16; msk <<= 1) tm = fmaxf(tm, __shfl_xor(tm, msk, 32));
      const float nm = fmaxf(rm[r], tm);
      alpha[r] = __expf(rm[r] - nm);
      const float p0 = __expf(s0 - nm);
      const float p1 = __expf(s1 - nm);
      float rs = p0 + p1;
#pragma unroll
      for (int msk = 1; msk < 16; msk <<= 1) rs += __shfl_xor(rs, msk, 32);
      rl[r] = rl[r] * alpha[r] + rs;
      rm[r] = nm;
      p0v[r] = p0; p1v[r] = p1;
    }
#pragma unroll
    for (int ct = 0; ct < 8; ++ct)
#pragma unroll
      for (int r = 0; r < 8; ++r) o[ct][r] *= alpha[r];

    // P: D-layout -> LDS -> A-fragment layout
    __syncthreads();
#pragma unroll
    for (int r = 0; r < 8; ++r) {
      ps[(r + 8 * half) * 32 + col]      = f2bf_raw(p0v[r]);
      ps[(r + 8 * half) * 32 + 16 + col] = f2bf_raw(p1v[r]);
    }
    __syncthreads();
    v16bf pf = load_a_frag(ps, 32, 0, 0, lane);

    // O += P . V  (B fragment: lane col = hd, element e -> key kt + e + 16*half)
#pragma unroll
    for (int ct = 0; ct < 8; ++ct) {
      v16bf vf;
#pragma unroll
      for (int e = 0; e < 16; ++e)
        vf[e] = raw2bf(Vb[(long)(kt + e + 16 * half) * HD_ + ct * 16 + col]);
      o[ct] = wmma_bf16(pf, vf, o[ct]);
    }
  }

  // epilogue: normalize and store head-interleaved [B,S,D]
#pragma unroll
  for (int ct = 0; ct < 8; ++ct)
#pragma unroll
    for (int r = 0; r < 8; ++r) {
      const int m = m0 + r + 8 * half;
      const float val = o[ct][r] / rl[r];
      O[((long)b * S_ + m) * D_ + h * HD_ + ct * 16 + col] = f2bf_raw(val);
    }
}

// ---------- 4) Output projection: [B*S,D] x [D,D]^T + bias -> f32 ----------
// grid: x = D/256, y = (B*S)/16 ; block = 128
__global__ __launch_bounds__(128) void out_gemm(const bfraw* __restrict__ A,
                                                const bfraw* __restrict__ W,
                                                const float* __restrict__ bias,
                                                float* __restrict__ C) {
  const int lane = threadIdx.x & 31;
  const int wave = threadIdx.x >> 5;
  const int half = lane >> 4;
  const int col  = lane & 15;
  const int m0   = blockIdx.y * 16;
  const int nw0  = (blockIdx.x * 4 + wave) * 64;

  v8f acc[4] = {V8F_ZERO, V8F_ZERO, V8F_ZERO, V8F_ZERO};
  for (int k0 = 0; k0 < D_; k0 += 32) {
    if (k0 + 32 < D_)
      __builtin_prefetch((const void*)(A + (long)(m0 + col) * D_ + k0 + 32), 0, 0);
    v16bf a = load_a_frag(A, D_, m0, k0, lane);
#pragma unroll
    for (int j = 0; j < 4; ++j) {
      v16bf b = load_b_frag_rowmajNK(W, D_, nw0 + j * 16, k0, lane);
      acc[j] = wmma_bf16(a, b, acc[j]);
    }
  }
#pragma unroll
  for (int j = 0; j < 4; ++j) {
    const int n = nw0 + j * 16 + col;
    const float bs = bias[n];
#pragma unroll
    for (int r = 0; r < 8; ++r) {
      const int m = m0 + r + 8 * half;
      C[(long)m * D_ + n] = acc[j][r] + bs;
    }
  }
}

// ---------- launch ----------
extern "C" void kernel_launch(void* const* d_in, const int* in_sizes, int n_in,
                              void* d_out, int out_size, void* d_ws, size_t ws_size,
                              hipStream_t stream) {
  (void)in_sizes; (void)n_in; (void)out_size; (void)ws_size;
  const float*          x      = (const float*)d_in[0];
  const float*          amask  = (const float*)d_in[1];
  const unsigned char*  kpad   = (const unsigned char*)d_in[2];  // jnp bool -> 1 byte
  const float*          qkv_w  = (const float*)d_in[3];
  const float*          qkv_b  = (const float*)d_in[4];
  const float*          out_w  = (const float*)d_in[5];
  const float*          out_b  = (const float*)d_in[6];
  float*                out    = (float*)d_out;

  char* ws = (char*)d_ws;
  size_t off = 0;
  auto alloc = [&](size_t bytes) -> char* {
    char* p = ws + off;
    off += (bytes + 255) & ~(size_t)255;
    return p;
  };
  const size_t nX   = (size_t)B_ * S_ * D_;        // 8.4M
  const size_t nWq  = (size_t)N3D * D_;            // 12.6M
  const size_t nWo  = (size_t)D_ * D_;             // 4.2M
  const size_t nHead = (size_t)B_ * H_ * S_ * HD_; // 8.4M

  bfraw* Xbf = (bfraw*)alloc(nX * 2);
  bfraw* Wq  = (bfraw*)alloc(nWq * 2);
  bfraw* Wo  = (bfraw*)alloc(nWo * 2);
  bfraw* Qb  = (bfraw*)alloc(nHead * 2);
  bfraw* Kb  = (bfraw*)alloc(nHead * 2);
  bfraw* Vb  = (bfraw*)alloc(nHead * 2);
  bfraw* Ob  = (bfraw*)alloc(nX * 2);

  cvt_f32_bf16<<<1024, 256, 0, stream>>>(x,     Xbf, (long)nX);
  cvt_f32_bf16<<<1024, 256, 0, stream>>>(qkv_w, Wq,  (long)nWq);
  cvt_f32_bf16<<<1024, 256, 0, stream>>>(out_w, Wo,  (long)nWo);

  qkv_gemm<<<dim3(N3D / 256, (B_ * S_) / 16), 128, 0, stream>>>(Xbf, Wq, qkv_b, Qb, Kb, Vb);
  flash_attn<<<dim3(S_ / 16, B_ * H_), 32, 0, stream>>>(Qb, Kb, Vb, amask, kpad, Ob);
  out_gemm<<<dim3(D_ / 256, (B_ * S_) / 16), 128, 0, stream>>>(Ob, Wo, out_b, out);
}
